// AutoregressiveDecoder_30648886624794
// MI455X (gfx1250) — matrix operand order repaired and checked by reading, not compiled, not run on hardware
//
#include <hip/hip_runtime.h>

typedef __attribute__((ext_vector_type(16))) __bf16 v16bf;
typedef __attribute__((ext_vector_type(8)))  float  v8f;

#define BB   256
#define NN   1000
#define DD   128
#define HH   8
#define STEPS 64
#define BN   (BB*NN)          // 256000 rows
#define NSWZ (24*4*32*16)     // 49152 pre-swizzled W_node elements
#define NEG_INF (-__builtin_inff())

// ---------------------------------------------------------------- utilities
__device__ __forceinline__ float bf16lo(unsigned int u) {
  union { unsigned int u; float f; } c; c.u = u << 16; return c.f;
}
__device__ __forceinline__ float bf16hi(unsigned int u) {
  union { unsigned int u; float f; } c; c.u = u & 0xffff0000u; return c.f;
}
__device__ __forceinline__ unsigned short f2bf_bits(float x) {
  union { __bf16 h; unsigned short u; } c; c.h = (__bf16)x; return c.u;
}

// ------------------------------------------------------- one-time kernels
// Pre-swizzle W_node[128 x 384] (f32) into per-lane WMMA-B fragment order:
// fragment (ct, kk), lane, elem e  ->  K = kk*32 + (e>>3)*16 + (lane>>4)*8 + (e&7),
//                                      col = ct*16 + (lane&15)
__global__ __launch_bounds__(256) void swizzle_wn_k(const float* __restrict__ wn,
                                                    __bf16* __restrict__ bswz) {
  int i = blockIdx.x * 256 + threadIdx.x;
  if (i >= NSWZ) return;
  int e    = i & 15;
  int lane = (i >> 4) & 31;
  int kk   = (i >> 9) & 3;
  int ct   = i >> 11;
  int half = lane >> 4, lr = lane & 15;
  int K   = kk * 32 + (e >> 3) * 16 + half * 8 + (e & 7);
  int col = ct * 16 + lr;
  bswz[i] = (__bf16)wn[K * (3 * DD) + col];
}

__global__ __launch_bounds__(256) void init_state_k(unsigned char* __restrict__ vis,
                                                    int* __restrict__ cur) {
  int i = blockIdx.x * 256 + threadIdx.x;
  if (i < BN) vis[i] = ((i % NN) == 0) ? 1 : 0;    // node 0 pre-visited
  if (i < BB) cur[i] = 0;                          // start at node 0
}

// graph_ctx[b] = mean_n(emb[b,n,:]) @ W_fixed
__global__ __launch_bounds__(128) void graph_ctx_k(const float* __restrict__ emb,
                                                   const float* __restrict__ wf,
                                                   float* __restrict__ gctx) {
  __shared__ float mean[DD];
  int b = blockIdx.x, d = threadIdx.x;
  const float* e = emb + (size_t)b * NN * DD;
  float s = 0.f;
  for (int n = 0; n < NN; ++n) s += e[n * DD + d];
  mean[d] = s * (1.0f / NN);
  __syncthreads();
  float acc = 0.f;
  for (int k = 0; k < DD; ++k) acc += mean[k] * wf[k * DD + d];
  gctx[b * DD + d] = acc;
}

// ------------------------------------------------ WMMA precompute GEMM
// C[256000 x 384] = emb[256000 x 128] * W_node[128 x 384], bf16 out,
// column-split into gk / gv / lk.  4 waves/block, 1 tile of 16 rows per
// wave, 24 column tiles, K=128 as 4 chained v_wmma_f32_16x16x32_bf16.
// A: 4 x b128 loads/chunk. B: one contiguous v16bf from pre-swizzled table.
// D: repacked via 512B LDS staging -> one b128 store per lane.
__global__ __launch_bounds__(128) void precompute_kvl_k(const float* __restrict__ emb,
                                                        const __bf16* __restrict__ bswz,
                                                        __bf16* __restrict__ kvl) {
  __shared__ __align__(16) unsigned short st[4][256];
  int tid  = threadIdx.x;
  int wave = tid >> 5;
  int lane = tid & 31;
  int half = lane >> 4;
  int lr   = lane & 15;
  long tile    = (long)blockIdx.x * 4 + wave;      // 16000 tiles total
  long rowbase = tile * 16;

  // A tiles (16x32 bf16): lane holds row M=lr;
  // elems 0..7 -> K = kk*32 + half*8 + e, elems 8..15 -> +16
  v16bf a[4];
  const float* arow = emb + (rowbase + lr) * (long)DD;
#pragma unroll
  for (int kk = 0; kk < 4; ++kk) {
    const float4 x0 = *(const float4*)(arow + kk * 32 + half * 8);
    const float4 x1 = *(const float4*)(arow + kk * 32 + half * 8 + 4);
    const float4 y0 = *(const float4*)(arow + kk * 32 + 16 + half * 8);
    const float4 y1 = *(const float4*)(arow + kk * 32 + 16 + half * 8 + 4);
    a[kk][0]  = (__bf16)x0.x; a[kk][1]  = (__bf16)x0.y;
    a[kk][2]  = (__bf16)x0.z; a[kk][3]  = (__bf16)x0.w;
    a[kk][4]  = (__bf16)x1.x; a[kk][5]  = (__bf16)x1.y;
    a[kk][6]  = (__bf16)x1.z; a[kk][7]  = (__bf16)x1.w;
    a[kk][8]  = (__bf16)y0.x; a[kk][9]  = (__bf16)y0.y;
    a[kk][10] = (__bf16)y0.z; a[kk][11] = (__bf16)y0.w;
    a[kk][12] = (__bf16)y1.x; a[kk][13] = (__bf16)y1.y;
    a[kk][14] = (__bf16)y1.z; a[kk][15] = (__bf16)y1.w;
  }

  const v16bf* bsw = (const v16bf*)bswz;
  unsigned short* kvb = (unsigned short*)kvl;

  for (int ct = 0; ct < 24; ++ct) {
    v8f c = {0.f, 0.f, 0.f, 0.f, 0.f, 0.f, 0.f, 0.f};
#pragma unroll
    for (int kk = 0; kk < 4; ++kk) {
      v16bf bm = bsw[(ct * 4 + kk) * 32 + lane];
      c = __builtin_amdgcn_wmma_f32_16x16x32_bf16(false, a[kk], false, bm,
                                                  (short)0, c, false, false);
    }
    // stage D tile (row = r + 8*half, col = lr) into LDS as bf16
#pragma unroll
    for (int r = 0; r < 8; ++r)
      st[wave][(r + 8 * half) * 16 + lr] = f2bf_bits(c[r]);
    __syncthreads();
    // coalesced store: each lane writes 8 contiguous bf16 (16B)
    {
      int row = lane >> 1;
      int seg = lane & 1;
      uint4 v = *(const uint4*)&st[wave][row * 16 + seg * 8];
      int  tgt = ct >> 3;                       // 0=gk 1=gv 2=lk
      int  lc  = (ct & 7) * 16 + seg * 8;       // local col in [0,128)
      long grow = rowbase + row;
      *(uint4*)(kvb + (size_t)tgt * BN * DD + grow * DD + lc) = v;
    }
    __syncthreads();
  }
}

// ------------------------------------------------ fused per-step kernel
// One block per batch: context query -> masked 8-head glimpse attention ->
// W_out projection -> logit matvec -> tanh clip -> masked log-softmax ->
// argmax -> state update.  256 threads.
__global__ __launch_bounds__(256) void step_k(const float* __restrict__ emb,
                                              const float* __restrict__ wc,
                                              const float* __restrict__ gctx,
                                              const float* __restrict__ wo,
                                              const __bf16* __restrict__ gk,
                                              const __bf16* __restrict__ gv,
                                              const __bf16* __restrict__ lkp,
                                              unsigned char* __restrict__ vis,
                                              int* __restrict__ cur,
                                              float* __restrict__ out_logp,
                                              float* __restrict__ out_act,
                                              int t) {
  __shared__ float cat[2 * DD];
  __shared__ float qs[DD];
  __shared__ float sc[HH][NN];
  __shared__ float red[HH][32];
  __shared__ float mh[HH], ish[HH];
  __shared__ float gl2[2][DD];
  __shared__ float gp[DD];
  __shared__ float l[NN];
  __shared__ float rv[256];
  __shared__ int   ri[256];

  int b = blockIdx.x, tid = threadIdx.x;
  const float* eb = emb + (size_t)b * NN * DD;
  unsigned char* vb = vis + (size_t)b * NN;
  int c = cur[b];

  // ---- context query: q = concat(first, cur) @ W_context + graph_ctx
  cat[tid] = (tid < DD) ? eb[tid] : eb[(size_t)c * DD + (tid - DD)];
  __syncthreads();
  if (tid < DD) {
    float acc = gctx[b * DD + tid];
    for (int k = 0; k < 2 * DD; ++k) acc += cat[k] * wc[k * DD + tid];
    qs[tid] = acc;
  }
  __syncthreads();

  // ---- glimpse scores for all heads
  for (int n = tid; n < NN; n += 256) {
    const unsigned int* kp = (const unsigned int*)(gk + ((size_t)b * NN + n) * DD);
    float acc[HH] = {0.f, 0.f, 0.f, 0.f, 0.f, 0.f, 0.f, 0.f};
#pragma unroll
    for (int dw = 0; dw < 64; ++dw) {
      unsigned int u = kp[dw];
      acc[dw >> 3] += qs[dw * 2] * bf16lo(u) + qs[dw * 2 + 1] * bf16hi(u);
    }
    bool m = vb[n] != 0;
#pragma unroll
    for (int h = 0; h < HH; ++h) sc[h][n] = m ? NEG_INF : acc[h] * 0.25f;  // 1/sqrt(16)
  }
  __syncthreads();

  // ---- per-head softmax stats (8 heads x 32 lanes)
  int h8 = tid >> 5, j = tid & 31;
  float pm = NEG_INF;
  for (int n = j; n < NN; n += 32) pm = fmaxf(pm, sc[h8][n]);
  red[h8][j] = pm;
  __syncthreads();
  if (tid < HH) {
    float m2 = NEG_INF;
    for (int k = 0; k < 32; ++k) m2 = fmaxf(m2, red[tid][k]);
    mh[tid] = m2;
  }
  __syncthreads();
  float mhv = mh[h8], ps = 0.f;
  for (int n = j; n < NN; n += 32) ps += expf(sc[h8][n] - mhv);
  red[h8][j] = ps;
  __syncthreads();
  if (tid < HH) {
    float s2 = 0.f;
    for (int k = 0; k < 32; ++k) s2 += red[tid][k];
    ish[tid] = 1.0f / s2;
  }
  __syncthreads();
  // normalize probabilities in place
  for (int n = tid; n < NN; n += 256) {
#pragma unroll
    for (int hh = 0; hh < HH; ++hh) sc[hh][n] = expf(sc[hh][n] - mh[hh]) * ish[hh];
  }
  __syncthreads();

  // ---- weighted sum over V: dim d = tid&127, node range split by tid>>7
  {
    int d = tid & 127, hs = tid >> 7;
    int h = d >> 4;
    int n0 = hs * 500, n1 = n0 + 500;
    const unsigned short* vp = (const unsigned short*)gv + (size_t)b * NN * DD + d;
    float acc = 0.f;
    for (int n = n0; n < n1; ++n) {
      union { unsigned int u; float f; } cvt;
      cvt.u = ((unsigned int)vp[(size_t)n * DD]) << 16;
      acc += sc[h][n] * cvt.f;
    }
    gl2[hs][d] = acc;
  }
  __syncthreads();
  // ---- W_out projection
  if (tid < DD) {
    float acc2 = 0.f;
    for (int k = 0; k < DD; ++k) acc2 += (gl2[0][k] + gl2[1][k]) * wo[k * DD + tid];
    gp[tid] = acc2;
  }
  __syncthreads();

  // ---- logits over logit-K + tanh clip + mask
  for (int n = tid; n < NN; n += 256) {
    const unsigned int* kp = (const unsigned int*)(lkp + ((size_t)b * NN + n) * DD);
    float acc = 0.f;
#pragma unroll
    for (int dw = 0; dw < 64; ++dw) {
      unsigned int u = kp[dw];
      acc += gp[dw * 2] * bf16lo(u) + gp[dw * 2 + 1] * bf16hi(u);
    }
    l[n] = vb[n] ? NEG_INF : 10.0f * tanhf(acc * 0.08838834764831845f); // 1/sqrt(128)
  }
  __syncthreads();

  // ---- max + argmax (lowest index on ties, matching jnp.argmax)
  float pmx = NEG_INF; int pix = 0;
  for (int n = tid; n < NN; n += 256)
    if (l[n] > pmx) { pmx = l[n]; pix = n; }
  rv[tid] = pmx; ri[tid] = pix;
  __syncthreads();
  for (int s = 128; s > 0; s >>= 1) {
    if (tid < s) {
      float ov = rv[tid + s]; int oi = ri[tid + s];
      if (ov > rv[tid] || (ov == rv[tid] && oi < ri[tid])) { rv[tid] = ov; ri[tid] = oi; }
    }
    __syncthreads();
  }
  float mv = rv[0]; int act = ri[0];
  __syncthreads();

  // ---- sum of exp
  float pse = 0.f;
  for (int n = tid; n < NN; n += 256) pse += expf(l[n] - mv);
  rv[tid] = pse;
  __syncthreads();
  for (int s = 128; s > 0; s >>= 1) {
    if (tid < s) rv[tid] += rv[tid + s];
    __syncthreads();
  }
  float lse = mv + logf(rv[0]);

  // ---- write log-probs + state update
  float* orow = out_logp + ((size_t)b * STEPS + t) * NN;
  for (int n = tid; n < NN; n += 256) orow[n] = l[n] - lse;
  if (tid == 0) {
    out_act[b * STEPS + t] = (float)act;
    vb[act] = 1;
    cur[b] = act;
  }
}

// ---------------------------------------------------------------- launcher
extern "C" void kernel_launch(void* const* d_in, const int* in_sizes, int n_in,
                              void* d_out, int out_size, void* d_ws, size_t ws_size,
                              hipStream_t stream) {
  (void)in_sizes; (void)n_in; (void)out_size; (void)ws_size;
  const float* emb = (const float*)d_in[0];   // [B,N,D]
  const float* wn  = (const float*)d_in[1];   // [D,3D]
  const float* wf  = (const float*)d_in[2];   // [D,D]
  const float* wc  = (const float*)d_in[3];   // [2D,D]
  const float* wo  = (const float*)d_in[4];   // [D,D]
  // d_in[5] = num_steps (device scalar) — fixed at 64 per reference.

  float* out_logp = (float*)d_out;                        // [B,64,N]
  float* out_act  = out_logp + (size_t)BB * STEPS * NN;   // [B,64] as f32

  // workspace layout (~197 MB)
  char* w = (char*)d_ws;
  size_t off = 0;
  __bf16* kvl  = (__bf16*)(w + off); off += (size_t)3 * BN * DD * sizeof(__bf16);
  off = (off + 255) & ~(size_t)255;
  __bf16* bswz = (__bf16*)(w + off); off += (size_t)NSWZ * sizeof(__bf16);
  off = (off + 255) & ~(size_t)255;
  float* gctx = (float*)(w + off); off += (size_t)BB * DD * sizeof(float);
  int*   cur  = (int*)(w + off);   off += (size_t)BB * sizeof(int);
  unsigned char* vis = (unsigned char*)(w + off);

  const __bf16* gk  = kvl;
  const __bf16* gv  = kvl + (size_t)BN * DD;
  const __bf16* lkp = kvl + (size_t)2 * BN * DD;

  swizzle_wn_k<<<(NSWZ + 255) / 256, 256, 0, stream>>>(wn, bswz);
  init_state_k<<<(BN + 255) / 256, 256, 0, stream>>>(vis, cur);
  graph_ctx_k<<<BB, 128, 0, stream>>>(emb, wf, gctx);
  precompute_kvl_k<<<BN / 16 / 4, 128, 0, stream>>>(emb, bswz, kvl);

  for (int t = 0; t < STEPS; ++t) {
    step_k<<<BB, 256, 0, stream>>>(emb, wc, gctx, wo, gk, gv, lkp,
                                   vis, cur, out_logp, out_act, t);
  }
}
